// Encoder_7842610283148
// MI455X (gfx1250) — matrix operand order repaired and checked by reading, not compiled, not run on hardware
//
#include <hip/hip_runtime.h>
#include <hip/hip_bf16.h>

// ---------------------------------------------------------------------------
// LayerNorm bidirectional GRU encoder for MI455X (gfx1250, wave32, WMMA).
//
//   1. convert/pad weights fp32 -> bf16
//   2. embedding gather -> xs bf16 [T,BK,320]
//   3. big WMMA GEMM: gi = xs @ Wih^T (both dirs)   [time-parallel]
//   4. LayerNorm(gi) -> bf16
//   5. persistent recurrence kernel (1 launch / layer): 32 WGs, each owns a
//      16-row batch slice + dir, loops T=128 steps; gh = h @ Whh^T via
//      v_wmma_f32_16x16x32_bf16; the next step's 48KB gi tile is streamed
//      into an LDS double buffer with global_load_async_to_lds_b128
//      (ASYNCcnt-tracked), overlapping the WMMA GEMM.
//   6. repeat 3-5 for layer 1 (input = y0 bf16)
//   7. output reduction kernels
// ---------------------------------------------------------------------------

#define T_SEQ 128
#define BK_   256
#define E_DIM 300
#define E_PAD 320
#define H_DIM 512
#define G_DIM 1536

typedef __attribute__((ext_vector_type(16))) __bf16 bf16x16;
typedef __attribute__((ext_vector_type(8)))  float  f32x8;

struct U32x8 { uint4 lo; uint4 hi; };   // 32 bytes == bf16x16

__device__ __forceinline__ unsigned short f2bf(float f) {
  unsigned u = __float_as_uint(f);
  return (unsigned short)((u + 0x7FFFu + ((u >> 16) & 1u)) >> 16);  // RNE
}
__device__ __forceinline__ float bf2f(unsigned short u) {
  return __uint_as_float(((unsigned)u) << 16);
}
__device__ __forceinline__ float sigm(float x) { return 1.f / (1.f + __expf(-x)); }

// A fragment (16x32 bf16, M x K). Lane L: row = L&15; holds K chunks
// [k0+8*(L>>4) .. +7] and [+16 .. +23]  (per ISA 16-bit A layout).
__device__ __forceinline__ bf16x16 load_a_frag(const unsigned short* A, int ld,
                                               int k0, int lane) {
  int m  = lane & 15;
  int kk = k0 + ((lane >> 4) << 3);
  const unsigned short* p = A + (size_t)m * ld + kk;
  U32x8 u;
  u.lo = *(const uint4*)(p);
  u.hi = *(const uint4*)(p + 16);
  return __builtin_bit_cast(bf16x16, u);
}

// B fragment (32x16 bf16, K x N) with B = W^T: column n of B == row (nb+n) of W.
// Lane L: n = L&15, holds K chunks [k0+8*(L>>4) .. +7] and [+16 .. +23].
__device__ __forceinline__ bf16x16 load_b_frag(const unsigned short* W, int ld,
                                               int nb, int k0, int lane) {
  int n  = lane & 15;
  int kk = k0 + ((lane >> 4) << 3);
  const unsigned short* p = W + (size_t)(nb + n) * ld + kk;
  U32x8 u;
  u.lo = *(const uint4*)(p);
  u.hi = *(const uint4*)(p + 16);
  return __builtin_bit_cast(bf16x16, u);
}

__device__ __forceinline__ f32x8 wmma_bf16(bf16x16 a, bf16x16 b, f32x8 c) {
  return __builtin_amdgcn_wmma_f32_16x16x32_bf16(false, a, false, b,
                                                 (short)0, c, false, false);
}

// ---------------------------------------------------------------------------
// fp32 -> bf16 weight convert with K-padding (E 300 -> 320 zero-fill)
__global__ void k_cvt_pad(const float* __restrict__ src,
                          unsigned short* __restrict__ dst,
                          int rows, int cols, int colsp) {
  size_t tot = (size_t)rows * colsp;
  for (size_t i = blockIdx.x * blockDim.x + threadIdx.x; i < tot;
       i += (size_t)gridDim.x * blockDim.x) {
    int r = (int)(i / colsp), c = (int)(i % colsp);
    dst[i] = (c < cols) ? f2bf(src[(size_t)r * cols + c]) : (unsigned short)0;
  }
}

// embedding gather -> xs bf16 [T][BK][E_PAD]
__global__ void k_embed(const int* __restrict__ src, const float* __restrict__ emb,
                        unsigned short* __restrict__ xs) {
  size_t tot = (size_t)T_SEQ * BK_ * E_PAD;
  for (size_t i = blockIdx.x * blockDim.x + threadIdx.x; i < tot;
       i += (size_t)gridDim.x * blockDim.x) {
    int kp = (int)(i % E_PAD);
    int tb = (int)(i / E_PAD);
    int b = tb % BK_, t = tb / BK_;
    float v = 0.f;
    if (kp < E_DIM) {
      int tok = src[(size_t)b * T_SEQ + t];
      v = emb[(size_t)tok * E_DIM + kp];
    }
    xs[i] = f2bf(v);
  }
}

// ---------------------------------------------------------------------------
// gi = A @ W^T   A:[Mtot,K] bf16, W:[2][G][K] bf16, C:[2][Mtot][G] fp32
__global__ void __launch_bounds__(256)
k_gemm_gi(const unsigned short* __restrict__ A,
          const unsigned short* __restrict__ W,
          float* __restrict__ C, int K) {
  int dir  = blockIdx.z;
  int m0   = blockIdx.y * 16;
  int wave = threadIdx.x >> 5, lane = threadIdx.x & 31;
  int n0   = blockIdx.x * 512 + wave * 64;

  const unsigned short* Ap = A + (size_t)m0 * K;
  const unsigned short* Wp = W + (size_t)dir * G_DIM * K;

  f32x8 acc[4];
#pragma unroll
  for (int q = 0; q < 4; q++)
#pragma unroll
    for (int i = 0; i < 8; i++) acc[q][i] = 0.f;

  for (int k0 = 0; k0 < K; k0 += 32) {
    bf16x16 a = load_a_frag(Ap, K, k0, lane);
#pragma unroll
    for (int q = 0; q < 4; q++) {
      bf16x16 b = load_b_frag(Wp, K, n0 + q * 16, k0, lane);
      acc[q] = wmma_bf16(a, b, acc[q]);
    }
  }

  float* Cp = C + ((size_t)dir * (T_SEQ * BK_) + m0) * G_DIM;
  int mrow = 8 * (lane >> 4), ncol = lane & 15;
#pragma unroll
  for (int q = 0; q < 4; q++)
#pragma unroll
    for (int i = 0; i < 8; i++)
      Cp[(size_t)(mrow + i) * G_DIM + n0 + q * 16 + ncol] = acc[q][i];
}

// row-wise LayerNorm over G=1536, affine, fp32 in -> bf16 out. wave32 per row.
__global__ void __launch_bounds__(256)
k_ln_gi(const float* __restrict__ gi, const float* __restrict__ gw,
        const float* __restrict__ bw, unsigned short* __restrict__ out) {
  int row  = blockIdx.x * 8 + (threadIdx.x >> 5);
  int lane = threadIdx.x & 31;
  int dir  = row / (T_SEQ * BK_);
  const float* g = gi + (size_t)row * G_DIM;
  float s = 0.f, s2 = 0.f;
  for (int j = lane; j < G_DIM; j += 32) { float v = g[j]; s += v; s2 += v * v; }
#pragma unroll
  for (int m = 16; m; m >>= 1) { s += __shfl_xor(s, m); s2 += __shfl_xor(s2, m); }
  float mean = s * (1.f / G_DIM);
  float rstd = rsqrtf(s2 * (1.f / G_DIM) - mean * mean + 1e-5f);
  const float* gwd = gw + (size_t)dir * G_DIM;
  const float* bwd = bw + (size_t)dir * G_DIM;
  unsigned short* o = out + (size_t)row * G_DIM;
  for (int j = lane; j < G_DIM; j += 32)
    o[j] = f2bf((g[j] - mean) * rstd * gwd[j] + bwd[j]);
}

// ---------------------------------------------------------------------------
// Persistent recurrence. LDS: ghs[16][1552] f32 | hf[16][512] f32 |
// hb[16][520] bf16 | gi double buffer 2 x 16x1536 bf16 (async-filled).
#define GHS_LD 1552
#define HB_LD  520
#define GI_TILE_BYTES (16 * G_DIM * 2)          // 49,152 B, fully contiguous

// Stream the 16x1536 bf16 gi tile for timestep t into LDS via the async
// global->LDS copy engine. 96 b128 instructions per WG = 12 per wave.
__device__ __forceinline__ void async_gi_prefetch(const unsigned short* gtile,
                                                  unsigned lds_base,
                                                  int wave, int lane) {
  unsigned idx = (unsigned)(wave * 12 * 32 + lane);
#pragma unroll
  for (int c = 0; c < 12; c++) {
    unsigned voff = (idx + (unsigned)c * 32u) * 16u;  // byte offset in tile
    unsigned ldsa = lds_base + voff;
    asm volatile("global_load_async_to_lds_b128 %0, %1, %2"
                 :
                 : "v"(ldsa), "v"(voff), "s"(gtile)
                 : "memory");
  }
}

__global__ void __launch_bounds__(256)
k_gru_layer(const unsigned short* __restrict__ gi_ln,  // [2][T][BK][G] bf16 (LN'd)
            const unsigned short* __restrict__ whh,    // [2][G][H] bf16
            const float* __restrict__ ghh, const float* __restrict__ bhh, // [2][G]
            const float* __restrict__ h0,              // [2][H] (this layer)
            float* __restrict__ y_f,                   // [T][BK][2H]
            unsigned short* __restrict__ y_b)          // same, bf16 (may be null)
{
  extern __shared__ char smem_raw[];
  float* ghs = (float*)smem_raw;                                   //  99,328 B
  float* hf  = (float*)(smem_raw + (size_t)16 * GHS_LD * 4);       //  32,768 B
  unsigned short* hb =
      (unsigned short*)(smem_raw + (size_t)16 * GHS_LD * 4 + 16 * H_DIM * 4);
  unsigned short* gibuf =                                          // 2x49,152 B
      (unsigned short*)((char*)hb + (size_t)16 * HB_LD * 2);
  unsigned gibuf_lds = (unsigned)(size_t)(void*)gibuf;  // flat->LDS addr[31:0]

  int dir  = blockIdx.y;
  int m0   = blockIdx.x * 16;
  int wave = threadIdx.x >> 5, lane = threadIdx.x & 31;

  // init h from broadcast h0
  const float* h0p = h0 + (size_t)dir * H_DIM;
  for (int i = threadIdx.x; i < 16 * H_DIM; i += 256) {
    int r = i >> 9, j = i & (H_DIM - 1);
    float v = h0p[j];
    hf[r * H_DIM + j] = v;
    hb[r * HB_LD + j] = f2bf(v);
  }

  const unsigned short* Wp = whh + (size_t)dir * G_DIM * H_DIM;
  const float* ghw = ghh + (size_t)dir * G_DIM;
  const float* bhw = bhh + (size_t)dir * G_DIM;
  int n0 = wave * 192;  // this wave's 12 N-tiles

  // kick off async fill of gi tile for step 0 into buffer 0
  {
    int t0 = dir ? (T_SEQ - 1) : 0;
    const unsigned short* gt =
        gi_ln + (((size_t)dir * T_SEQ + t0) * BK_ + m0) * G_DIM;
    async_gi_prefetch(gt, gibuf_lds, wave, lane);
  }
  __syncthreads();

  for (int s = 0; s < T_SEQ; s++) {
    int t = dir ? (T_SEQ - 1 - s) : s;

    // async prefetch of next step's gi tile, overlapped with the GEMM below
    if (s + 1 < T_SEQ) {
      int tn = dir ? (T_SEQ - 2 - s) : (s + 1);
      const unsigned short* gt =
          gi_ln + (((size_t)dir * T_SEQ + tn) * BK_ + m0) * G_DIM;
      async_gi_prefetch(gt, gibuf_lds + ((s + 1) & 1) * GI_TILE_BYTES,
                        wave, lane);
    }

    // ---- phase 1: gh = h @ Whh^T  (16 x 1536, K=512) ----
    f32x8 acc[12];
#pragma unroll
    for (int q = 0; q < 12; q++)
#pragma unroll
      for (int i = 0; i < 8; i++) acc[q][i] = 0.f;

    for (int k0 = 0; k0 < H_DIM; k0 += 32) {
      bf16x16 a = load_a_frag(hb, HB_LD, k0, lane);   // from LDS
#pragma unroll
      for (int q = 0; q < 12; q++) {
        bf16x16 b = load_b_frag(Wp, H_DIM, n0 + q * 16, k0, lane);
        acc[q] = wmma_bf16(a, b, acc[q]);
      }
    }
    int mrow = 8 * (lane >> 4), ncol = lane & 15;
#pragma unroll
    for (int q = 0; q < 12; q++)
#pragma unroll
      for (int i = 0; i < 8; i++)
        ghs[(mrow + i) * GHS_LD + n0 + q * 16 + ncol] = acc[q][i];

    // Async loads retire in order: after issuing 12 new ones for step s+1,
    // ASYNCcnt<=12 guarantees step s's 12 are done. Last step: wait for 0.
    if (s + 1 < T_SEQ) {
      asm volatile("s_wait_asynccnt 0xc" ::: "memory");
    } else {
      asm volatile("s_wait_asynccnt 0x0" ::: "memory");
    }
    __syncthreads();   // publishes ghs stores + all waves' async LDS fills

    const unsigned short* gis = gibuf + (size_t)(s & 1) * (16 * G_DIM);

    // ---- phase 2: LN(gh) + gates + h update; wave w -> rows 2w, 2w+1 ----
    for (int rr = 0; rr < 2; rr++) {
      int row = wave * 2 + rr;
      int b   = m0 + row;
      const float* gr = ghs + (size_t)row * GHS_LD;
      float s1 = 0.f, s2 = 0.f;
      for (int j = lane; j < G_DIM; j += 32) { float v = gr[j]; s1 += v; s2 += v * v; }
#pragma unroll
      for (int m = 16; m; m >>= 1) { s1 += __shfl_xor(s1, m); s2 += __shfl_xor(s2, m); }
      float mean = s1 * (1.f / G_DIM);
      float rstd = rsqrtf(s2 * (1.f / G_DIM) - mean * mean + 1e-5f);

      const unsigned short* gi = gis + (size_t)row * G_DIM;  // LDS, async-filled
      for (int j = lane; j < H_DIM; j += 32) {
        float hr = (gr[j]        - mean) * rstd * ghw[j]        + bhw[j];
        float hz = (gr[j + 512]  - mean) * rstd * ghw[j + 512]  + bhw[j + 512];
        float hn = (gr[j + 1024] - mean) * rstd * ghw[j + 1024] + bhw[j + 1024];
        float ir = bf2f(gi[j]), iz = bf2f(gi[j + 512]), inn = bf2f(gi[j + 1024]);
        float r = sigm(ir + hr);
        float z = sigm(iz + hz);
        float n = tanhf(inn + r * hn);
        float hprev = hf[row * H_DIM + j];
        float hnew  = (1.f - z) * n + z * hprev;
        hf[row * H_DIM + j] = hnew;
        hb[row * HB_LD + j] = f2bf(hnew);
        size_t yi = (((size_t)t * BK_ + b) * 2 * H_DIM) + (size_t)dir * H_DIM + j;
        y_f[yi] = hnew;
        if (y_b) y_b[yi] = f2bf(hnew);
      }
    }
    __syncthreads();
  }
}

// ---------------------------------------------------------------------------
// out1[b][t][j] = mean over {layer0,layer1} x {fwd,bwd}
__global__ void k_out1(const float* __restrict__ y0, const float* __restrict__ y1,
                       float* __restrict__ out1) {
  size_t tot = (size_t)BK_ * T_SEQ * H_DIM;
  for (size_t i = blockIdx.x * blockDim.x + threadIdx.x; i < tot;
       i += (size_t)gridDim.x * blockDim.x) {
    int j = (int)(i % H_DIM);
    int t = (int)((i / H_DIM) % T_SEQ);
    int b = (int)(i / ((size_t)H_DIM * T_SEQ));
    size_t base = ((size_t)t * BK_ + b) * (2 * H_DIM);
    out1[i] = 0.25f * (y0[base + j] + y0[base + H_DIM + j] +
                       y1[base + j] + y1[base + H_DIM + j]);
  }
}

// out2[b][j] = ( hidden[b,0,bwd,j] + hidden[b,lengths[b],fwd,j] ) / 2
__global__ void k_out2(const float* __restrict__ y0, const float* __restrict__ y1,
                       const int* __restrict__ lengths, float* __restrict__ out2) {
  size_t tot = (size_t)BK_ * H_DIM;
  for (size_t i = blockIdx.x * blockDim.x + threadIdx.x; i < tot;
       i += (size_t)gridDim.x * blockDim.x) {
    int b = (int)(i / H_DIM), j = (int)(i % H_DIM);
    int L = lengths[b];
    size_t b0 = ((size_t)0 * BK_ + b) * (2 * H_DIM);
    size_t bL = ((size_t)L * BK_ + b) * (2 * H_DIM);
    float hb0 = 0.5f * (y0[b0 + H_DIM + j] + y1[b0 + H_DIM + j]);  // t=0, bwd
    float hfL = 0.5f * (y0[bL + j]          + y1[bL + j]);          // t=L, fwd
    out2[i] = 0.5f * (hb0 + hfL);
  }
}

// ---------------------------------------------------------------------------
extern "C" void kernel_launch(void* const* d_in, const int* in_sizes, int n_in,
                              void* d_out, int out_size, void* d_ws, size_t ws_size,
                              hipStream_t stream) {
  (void)in_sizes; (void)n_in; (void)out_size; (void)ws_size;
  const int*   src     = (const int*)  d_in[0];
  const int*   lengths = (const int*)  d_in[1];
  const float* emb     = (const float*)d_in[2];
  const float* Wih0    = (const float*)d_in[3];
  const float* Whh0    = (const float*)d_in[4];
  const float* gih0    = (const float*)d_in[5];
  const float* bih0    = (const float*)d_in[6];
  const float* ghh0    = (const float*)d_in[7];
  const float* bhh0    = (const float*)d_in[8];
  const float* Wih1    = (const float*)d_in[9];
  const float* Whh1    = (const float*)d_in[10];
  const float* gih1    = (const float*)d_in[11];
  const float* bih1    = (const float*)d_in[12];
  const float* ghh1    = (const float*)d_in[13];
  const float* bhh1    = (const float*)d_in[14];
  const float* h0      = (const float*)d_in[15];
  float* out = (float*)d_out;

  // workspace carve (256B aligned)
  char* ws = (char*)d_ws;
  size_t off = 0;
  auto carve = [&](size_t bytes) -> char* {
    char* p = ws + off;
    off += (bytes + 255) & ~(size_t)255;
    return p;
  };
  unsigned short* wih0_b = (unsigned short*)carve((size_t)2 * G_DIM * E_PAD * 2);
  unsigned short* whh0_b = (unsigned short*)carve((size_t)2 * G_DIM * H_DIM * 2);
  unsigned short* wih1_b = (unsigned short*)carve((size_t)2 * G_DIM * 1024 * 2);
  unsigned short* whh1_b = (unsigned short*)carve((size_t)2 * G_DIM * H_DIM * 2);
  unsigned short* xs_b   = (unsigned short*)carve((size_t)T_SEQ * BK_ * E_PAD * 2);
  float*          y0_f   = (float*)carve((size_t)T_SEQ * BK_ * 1024 * 4);
  unsigned short* y0_b   = (unsigned short*)carve((size_t)T_SEQ * BK_ * 1024 * 2);
  float*          y1_f   = (float*)carve((size_t)T_SEQ * BK_ * 1024 * 4);
  float*          gi_raw = (float*)carve((size_t)2 * T_SEQ * BK_ * G_DIM * 4);
  unsigned short* gi_ln  = (unsigned short*)carve((size_t)2 * T_SEQ * BK_ * G_DIM * 2);

  // 1. weight conversion (bf16, pad E to 320)
  k_cvt_pad<<<2048, 256, 0, stream>>>(Wih0, wih0_b, 2 * G_DIM, E_DIM, E_PAD);
  k_cvt_pad<<<2048, 256, 0, stream>>>(Whh0, whh0_b, 2 * G_DIM, H_DIM, H_DIM);
  k_cvt_pad<<<4096, 256, 0, stream>>>(Wih1, wih1_b, 2 * G_DIM, 1024, 1024);
  k_cvt_pad<<<2048, 256, 0, stream>>>(Whh1, whh1_b, 2 * G_DIM, H_DIM, H_DIM);

  // 2. embedding gather
  k_embed<<<8192, 256, 0, stream>>>(src, emb, xs_b);

  dim3 gemm_grid(G_DIM / 512, (T_SEQ * BK_) / 16, 2);
  int ln_blocks = (2 * T_SEQ * BK_) / 8;
  size_t shmem = (size_t)16 * GHS_LD * 4 + (size_t)16 * H_DIM * 4 +
                 (size_t)16 * HB_LD * 2 + 2 * (size_t)GI_TILE_BYTES; // 247,040 B
  (void)hipFuncSetAttribute((const void*)k_gru_layer,
                            hipFuncAttributeMaxDynamicSharedMemorySize,
                            (int)shmem);

  // 3-5. layer 0
  k_gemm_gi<<<gemm_grid, 256, 0, stream>>>(xs_b, wih0_b, gi_raw, E_PAD);
  k_ln_gi<<<ln_blocks, 256, 0, stream>>>(gi_raw, gih0, bih0, gi_ln);
  k_gru_layer<<<dim3(16, 2), 256, shmem, stream>>>(gi_ln, whh0_b, ghh0, bhh0,
                                                   h0 + 0, y0_f, y0_b);
  // 6. layer 1 (input = y0 bf16)
  k_gemm_gi<<<gemm_grid, 256, 0, stream>>>(y0_b, wih1_b, gi_raw, 1024);
  k_ln_gi<<<ln_blocks, 256, 0, stream>>>(gi_raw, gih1, bih1, gi_ln);
  k_gru_layer<<<dim3(16, 2), 256, shmem, stream>>>(gi_ln, whh1_b, ghh1, bhh1,
                                                   h0 + 2 * H_DIM, y1_f, nullptr);

  // 7. outputs
  k_out1<<<16384, 256, 0, stream>>>(y0_f, y1_f, out);
  k_out2<<<512, 256, 0, stream>>>(y0_f, y1_f, lengths,
                                  out + (size_t)BK_ * T_SEQ * H_DIM);
}
